// MPNEncoder_52432960749757
// MI455X (gfx1250) — matrix operand order repaired and checked by reading, not compiled, not run on hardware
//
#include <hip/hip_runtime.h>
#include <hip/hip_bf16.h>
#include <math.h>

// ---------------------------------------------------------------------------
// MPN encoder for MI455X (gfx1250, wave32, WMMA).
// All GEMMs use v_wmma_f32_16x16x32_bf16 with fp32 accumulation.
// ---------------------------------------------------------------------------

typedef __attribute__((ext_vector_type(16))) __bf16 v16bf;
typedef __attribute__((ext_vector_type(8)))  float  v8f;

#define MATOMS   50000
#define NBONDS   100000
#define HID      300
#define AFD      133
#define BFD      14
#define MAXNB    6
#define PERMOL   25
#define NMOLS    (MATOMS / PERMOL)
#define MTILES   (MATOMS / 16)          // 3125, exact
#define INV_SCALE 0.05773502691896258f  // 1/sqrt(300)
#define INV_H     (1.0f / 300.0f)
#define LN_EPS    1e-5f

// ---------------------------------------------------------------------------
// Weight packing: fp32 [K,N] (row-major, optional row offset) -> bf16 blobs in
// the exact B-fragment layout for v_wmma_f32_16x16x32_bf16:
//   blob[(kt*ntiles + nt)*512 + lane*16 + e]
//   lanes 0..15 : column n0+lane,     K = k0 + {0..7, 16..23}
//   lanes 16..31: column n0+lane-16,  K = k0 + {8..15, 24..31}
// Out-of-range K/N -> 0 (zero padding makes padded K-tiles exact).
// ---------------------------------------------------------------------------
__global__ void k_pack_w(const float* __restrict__ W, int row0, int K, int N,
                         int ldw, int ktiles, int ntiles, __bf16* __restrict__ out)
{
    size_t idx = (size_t)blockIdx.x * blockDim.x + threadIdx.x;
    size_t total = (size_t)ktiles * ntiles * 512;
    if (idx >= total) return;
    int e    = (int)(idx & 15);
    int lane = (int)((idx >> 4) & 31);
    size_t tile = idx >> 9;
    int nt = (int)(tile % ntiles);
    int kt = (int)(tile / ntiles);
    int k = kt * 32 + ((lane < 16) ? 0 : 8) + ((e < 8) ? e : e + 8);
    int n = nt * 16 + (lane & 15);
    float v = (k < K && n < N) ? W[(size_t)(row0 + k) * ldw + n] : 0.0f;
    out[idx] = (__bf16)v;
}

// fp32 [M,K] -> bf16 [M,Kp] (zero padded)
__global__ void k_cvt_pad(const float* __restrict__ src, __bf16* __restrict__ dst,
                          int M, int K, int Kp)
{
    size_t idx = (size_t)blockIdx.x * blockDim.x + threadIdx.x;
    if (idx >= (size_t)M * Kp) return;
    int kp = (int)(idx % Kp);
    size_t row = idx / Kp;
    float v = (kp < K) ? src[row * (size_t)K + kp] : 0.0f;
    dst[idx] = (__bf16)v;
}

__global__ void k_relu(const float* __restrict__ s, float* __restrict__ d, size_t n)
{
    size_t i = (size_t)blockIdx.x * blockDim.x + threadIdx.x;
    if (i < n) d[i] = fmaxf(s[i], 0.0f);
}

// Wqk[m,i] = sum_j Wq[m,j]*Wk[i,j]  (i<300 -> Wqk, i>=300 -> Wqb col i-300)
__global__ void k_make_wqk(const float* __restrict__ Wq, const float* __restrict__ Wk,
                           float* __restrict__ Wqk, float* __restrict__ Wqb)
{
    int idx = blockIdx.x * blockDim.x + threadIdx.x;
    if (idx >= HID * (HID + BFD)) return;
    int i = idx % (HID + BFD);
    int m = idx / (HID + BFD);
    const float* wq = Wq + (size_t)m * HID;
    const float* wk = Wk + (size_t)i * HID;
    float s = 0.0f;
    for (int j = 0; j < HID; ++j) s += wq[j] * wk[j];
    if (i < HID) Wqk[(size_t)m * HID + i] = s;
    else         Wqb[(size_t)m * BFD + (i - HID)] = s;
}

// ---------------------------------------------------------------------------
// WMMA GEMM: C[M,N] = act(A_bf16[M,Kp] @ Bpacked + bias + Cin)
// wave -> 16(M) x NV*16(N) tile, 8 waves/block. NV is a template constant so
// the K loop contains zero branches; A/B fragments are double-buffered so the
// next K-tile's loads overlap the current WMMAs (partial s_wait_loadcnt).
// A fragment layout (16-bit A 16x32):
//   lanes 0..15 : row m0+lane,    K in {0..7, 16..23}
//   lanes 16..31: row m0+lane-16, K in {8..15, 24..31}
// ---------------------------------------------------------------------------
union ABu { v16bf v; float4 q[2]; };

template <int NV>
__global__ __launch_bounds__(256)
void k_gemm_t(const __bf16* __restrict__ A, const __bf16* __restrict__ Bp,
              float* __restrict__ C, const float* __restrict__ bias,
              const float* __restrict__ Cin, int M, int N,
              int ktiles, int ntiles, int nt_base, int relu)
{
    const int lane  = threadIdx.x & 31;
    const int wave  = threadIdx.x >> 5;
    const int mtile = blockIdx.x * 8 + wave;
    if (mtile * 16 >= M) return;                 // wave-uniform guard
    const int nt0 = nt_base + blockIdx.y * 4;    // all NV tiles valid by dispatch
    const int Kp  = ktiles * 32;
    const int row = mtile * 16 + (lane & 15);
    const int koff = (lane < 16) ? 0 : 8;
    const __bf16* arow = A + (size_t)row * Kp + koff;
    const __bf16* brow = Bp + ((size_t)nt0 << 9) + (size_t)lane * 16;
    const size_t  bstride = (size_t)ntiles << 9;     // bf16 elems per K-tile

    v8f acc[NV] = {};

    auto loadA = [&](int kt, ABu& a) {
        const float4* ap = reinterpret_cast<const float4*>(arow + kt * 32);
        a.q[0] = ap[0];      // K elems e=0..7
        a.q[1] = ap[2];      // K elems e=8..15 (offset +16 bf16)
    };
    auto loadB = [&](int kt, int t, ABu& b) {
        const float4* bp = reinterpret_cast<const float4*>(
            brow + (size_t)kt * bstride + ((size_t)t << 9));
        b.q[0] = bp[0];
        b.q[1] = bp[1];
    };

    ABu a0, b0[NV];
    loadA(0, a0);
#pragma unroll
    for (int t = 0; t < NV; ++t) loadB(0, t, b0[t]);

    for (int kt = 0; kt < ktiles; ++kt) {
        ABu a1, b1[NV];
        const int ktn = (kt + 1 < ktiles) ? kt + 1 : kt;  // harmless reload on last iter
        loadA(ktn, a1);
#pragma unroll
        for (int t = 0; t < NV; ++t) loadB(ktn, t, b1[t]);
#pragma unroll
        for (int t = 0; t < NV; ++t)
            acc[t] = __builtin_amdgcn_wmma_f32_16x16x32_bf16(
                false, a0.v, false, b0[t].v, (short)0, acc[t], false, false);
        a0 = a1;
#pragma unroll
        for (int t = 0; t < NV; ++t) b0[t] = b1[t];
    }

    // C/D layout: VGPR r, lanes 0..15 -> M=r, lanes 16..31 -> M=r+8; N = lane&15
    const int crow = mtile * 16 + ((lane >= 16) ? 8 : 0);
    const int ccol = lane & 15;
#pragma unroll
    for (int t = 0; t < NV; ++t) {
        int cc = (nt0 + t) * 16 + ccol;
        if (cc >= N) continue;
        float bval = bias ? bias[cc] : 0.0f;
#pragma unroll
        for (int r = 0; r < 8; ++r) {
            size_t idx = (size_t)(crow + r) * N + cc;
            float val = acc[t][r] + bval;
            if (Cin)  val += Cin[idx];
            if (relu) val = fmaxf(val, 0.0f);
            C[idx] = val;
        }
    }
}

// ---------------------------------------------------------------------------
// Attention + ReLU + LayerNorm, one wave per atom (wave32 reductions).
// score[n,k] = (message[a2a[n,k]] . qk[n]) + (f_bonds[a2b[n,k]] . qb[n])
// mw = LN(relu(inp + softmax(score/sqrt(H)) . v[a2a]))
// ---------------------------------------------------------------------------
__global__ __launch_bounds__(256)
void k_attn(const float* __restrict__ message, const float* __restrict__ qk,
            const float* __restrict__ qb, const float* __restrict__ v,
            const float* __restrict__ inp, const float* __restrict__ f_bonds,
            const int* __restrict__ a2a, const int* __restrict__ a2b,
            const float* __restrict__ ln_g, const float* __restrict__ ln_b,
            float* __restrict__ mw)
{
    const int lane = threadIdx.x & 31;
    const int n = blockIdx.x * 8 + (threadIdx.x >> 5);
    if (n >= MATOMS) return;

    float qkr[10];
    for (int t = 0; t < 10; ++t) {
        int c = lane + t * 32;
        qkr[t] = (c < HID) ? qk[(size_t)n * HID + c] : 0.0f;
    }
    float qbl = (lane < BFD) ? qb[(size_t)n * BFD + lane] : 0.0f;

    int   nbr[MAXNB];
    float sc[MAXNB];
    for (int k = 0; k < MAXNB; ++k) {
        int na = a2a[(size_t)n * MAXNB + k];
        int nb = a2b[(size_t)n * MAXNB + k];
        nbr[k] = na;
        const float* mrow = message + (size_t)na * HID;
        float s = 0.0f;
        for (int t = 0; t < 10; ++t) {
            int c = lane + t * 32;
            if (c < HID) s += mrow[c] * qkr[t];
        }
        if (lane < BFD) s += f_bonds[(size_t)nb * BFD + lane] * qbl;
        for (int off = 16; off; off >>= 1) s += __shfl_xor(s, off);
        sc[k] = s * INV_SCALE;
    }

    // softmax over 6 (redundant per lane; all lanes hold identical reduced vals)
    float mx = sc[0];
    for (int k = 1; k < MAXNB; ++k) mx = fmaxf(mx, sc[k]);
    float ew[MAXNB], den = 0.0f;
    for (int k = 0; k < MAXNB; ++k) { ew[k] = __expf(sc[k] - mx); den += ew[k]; }
    float inv = 1.0f / den;

    float mwr[10], sum = 0.0f, sumsq = 0.0f;
    for (int t = 0; t < 10; ++t) {
        int c = lane + t * 32;
        float a = 0.0f;
        if (c < HID) {
            for (int k = 0; k < MAXNB; ++k)
                a += ew[k] * v[(size_t)nbr[k] * HID + c];
            a = a * inv + inp[(size_t)n * HID + c];
            a = fmaxf(a, 0.0f);
        }
        mwr[t] = a;
        sum += a;
        sumsq += a * a;
    }
    for (int off = 16; off; off >>= 1) {
        sum   += __shfl_xor(sum, off);
        sumsq += __shfl_xor(sumsq, off);
    }
    float mu  = sum * INV_H;
    float var = sumsq * INV_H - mu * mu;
    float rstd = rsqrtf(var + LN_EPS);
    for (int t = 0; t < 10; ++t) {
        int c = lane + t * 32;
        if (c < HID)
            mw[(size_t)n * HID + c] = (mwr[t] - mu) * rstd * ln_g[c] + ln_b[c];
    }
}

__device__ __forceinline__ float sigmoidf_(float x) { return 1.0f / (1.0f + __expf(-x)); }

// GRU elementwise: gi/gh already include biases. h updated in place.
__global__ void k_gru(const float* __restrict__ gi, const float* __restrict__ gh,
                      float* __restrict__ h)
{
    size_t idx = (size_t)blockIdx.x * blockDim.x + threadIdx.x;
    if (idx >= (size_t)MATOMS * HID) return;
    int j = (int)(idx % HID);
    size_t row = idx / HID;
    const float* gir = gi + row * (size_t)(3 * HID);
    const float* ghr = gh + row * (size_t)(3 * HID);
    float r  = sigmoidf_(gir[j] + ghr[j]);
    float z  = sigmoidf_(gir[HID + j] + ghr[HID + j]);
    float nn = tanhf(gir[2 * HID + j] + r * ghr[2 * HID + j]);
    float hv = h[idx];
    h[idx] = (1.0f - z) * nn + z * hv;
}

__global__ void k_gather_sum(const float* __restrict__ message,
                             const int* __restrict__ a2a, float* __restrict__ amsg)
{
    size_t idx = (size_t)blockIdx.x * blockDim.x + threadIdx.x;
    if (idx >= (size_t)MATOMS * HID) return;
    int c = (int)(idx % HID);
    size_t n = idx / HID;
    const int* arow = a2a + n * MAXNB;
    float s = 0.0f;
    for (int k = 0; k < MAXNB; ++k)
        s += message[(size_t)arow[k] * HID + c];
    amsg[idx] = s;
}

__global__ void k_segmean(const float* __restrict__ ah, float* __restrict__ out)
{
    size_t idx = (size_t)blockIdx.x * blockDim.x + threadIdx.x;
    if (idx >= (size_t)NMOLS * HID) return;
    int c = (int)(idx % HID);
    size_t m = idx / HID;
    float s = 0.0f;
    for (int a = 0; a < PERMOL; ++a)
        s += ah[((size_t)m * PERMOL + a) * HID + c];
    out[idx] = s * (1.0f / PERMOL);
}

// ---------------------------------------------------------------------------
// Host side
// ---------------------------------------------------------------------------
static void gemm_launch(const __bf16* A, const __bf16* Bp, float* C,
                        const float* bias, const float* Cin,
                        int N, int ktiles, int ntiles, int relu, hipStream_t s)
{
    const unsigned gx = (MTILES + 7) / 8;
    const int full = ntiles / 4;
    const int rem  = ntiles % 4;
    if (full > 0)
        k_gemm_t<4><<<dim3(gx, full), 256, 0, s>>>(A, Bp, C, bias, Cin,
                                                   MATOMS, N, ktiles, ntiles, 0, relu);
    if (rem) {
        const int base = full * 4;
        dim3 g(gx, 1);
        switch (rem) {
        case 1: k_gemm_t<1><<<g, 256, 0, s>>>(A, Bp, C, bias, Cin, MATOMS, N, ktiles, ntiles, base, relu); break;
        case 2: k_gemm_t<2><<<g, 256, 0, s>>>(A, Bp, C, bias, Cin, MATOMS, N, ktiles, ntiles, base, relu); break;
        default:k_gemm_t<3><<<g, 256, 0, s>>>(A, Bp, C, bias, Cin, MATOMS, N, ktiles, ntiles, base, relu); break;
        }
    }
}

static inline size_t ew_blocks(size_t n) { return (n + 255) / 256; }

extern "C" void kernel_launch(void* const* d_in, const int* in_sizes, int n_in,
                              void* d_out, int out_size, void* d_ws, size_t ws_size,
                              hipStream_t stream)
{
    (void)in_sizes; (void)n_in; (void)out_size; (void)ws_size;

    const float* f_atoms = (const float*)d_in[0];
    const float* f_bonds = (const float*)d_in[1];
    const int*   a2a     = (const int*)d_in[2];
    const int*   a2b     = (const int*)d_in[3];
    const float* W_i     = (const float*)d_in[4];
    const float* Wk      = (const float*)d_in[5];
    const float* Wq      = (const float*)d_in[6];
    const float* Wv      = (const float*)d_in[7];
    const float* ln_g    = (const float*)d_in[8];
    const float* ln_b    = (const float*)d_in[9];
    const float* gru_wih = (const float*)d_in[10];
    const float* gru_whh = (const float*)d_in[11];
    const float* gru_bih = (const float*)d_in[12];
    const float* gru_bhh = (const float*)d_in[13];
    const float* W_o     = (const float*)d_in[14];
    const float* b_o     = (const float*)d_in[15];
    float* out = (float*)d_out;

    // -------- workspace layout (bump allocator, 256B aligned) --------
    char* ws = (char*)d_ws;
    size_t cur = 0;
    auto alloc = [&](size_t bytes) { size_t r = cur; cur += (bytes + 255) & ~(size_t)255; return r; };

    const size_t MH   = (size_t)MATOMS * HID;
    float*  inp    = (float*)(ws + alloc(MH * 4));
    float*  msg    = (float*)(ws + alloc(MH * 4));
    __bf16* msgbf  = (__bf16*)(ws + alloc((size_t)MATOMS * 320 * 2));
    __bf16* fabf   = (__bf16*)(ws + alloc((size_t)MATOMS * 160 * 2));
    float*  mwbuf  = (float*)(ws + alloc(MH * 4));
    __bf16* mwbf   = (__bf16*)(ws + alloc((size_t)MATOMS * 320 * 2));
    // big region: {qk | v | qb} early, {gi | gh} during GRU (qk/v/qb dead then)
    size_t bigoff  = alloc(2 * (size_t)MATOMS * 900 * 4);
    float* qk = (float*)(ws + bigoff);
    float* vv = qk + MH;
    float* qb = vv + MH;
    float* gi = (float*)(ws + bigoff);
    float* gh = gi + (size_t)MATOMS * 900;
    float* t_buf  = qk;     // reuse at readout
    float* ah     = vv;     // reuse at readout
    float* amsg   = mwbuf;  // reuse at readout
    __bf16* amsgbf = mwbf;

    float*  WqkF = (float*)(ws + alloc((size_t)HID * HID * 4));
    float*  WqbF = (float*)(ws + alloc((size_t)HID * BFD * 4));
    __bf16* WiP  = (__bf16*)(ws + alloc((size_t)5  * 19 * 512 * 2));
    __bf16* WqkP = (__bf16*)(ws + alloc((size_t)10 * 19 * 512 * 2));
    __bf16* WqbP = (__bf16*)(ws + alloc((size_t)10 * 1  * 512 * 2));
    __bf16* WvP  = (__bf16*)(ws + alloc((size_t)10 * 19 * 512 * 2));
    __bf16* WihP = (__bf16*)(ws + alloc((size_t)10 * 57 * 512 * 2));
    __bf16* WhhP = (__bf16*)(ws + alloc((size_t)10 * 57 * 512 * 2));
    __bf16* WoTP = (__bf16*)(ws + alloc((size_t)5  * 19 * 512 * 2));
    __bf16* WoBP = (__bf16*)(ws + alloc((size_t)10 * 19 * 512 * 2));

    // -------- weight preparation --------
    k_make_wqk<<<(HID * (HID + BFD) + 255) / 256, 256, 0, stream>>>(Wq, Wk, WqkF, WqbF);
    auto pack = [&](const float* W, int row0, int K, int N, int ldw, int kt, int nt, __bf16* dst) {
        size_t tot = (size_t)kt * nt * 512;
        k_pack_w<<<(unsigned)ew_blocks(tot), 256, 0, stream>>>(W, row0, K, N, ldw, kt, nt, dst);
    };
    pack(W_i,     0,   AFD, HID,     HID,     5,  19, WiP);
    pack(WqkF,    0,   HID, HID,     HID,     10, 19, WqkP);
    pack(WqbF,    0,   HID, BFD,     BFD,     10, 1,  WqbP);
    pack(Wv,      0,   HID, HID,     HID,     10, 19, WvP);
    pack(gru_wih, 0,   HID, 3 * HID, 3 * HID, 10, 57, WihP);
    pack(gru_whh, 0,   HID, 3 * HID, 3 * HID, 10, 57, WhhP);
    pack(W_o,     0,   AFD, HID,     HID,     5,  19, WoTP);
    pack(W_o,     AFD, HID, HID,     HID,     10, 19, WoBP);

    // -------- input projection --------
    k_cvt_pad<<<(unsigned)ew_blocks((size_t)MATOMS * 160), 256, 0, stream>>>(f_atoms, fabf, MATOMS, AFD, 160);
    gemm_launch(fabf, WiP, inp, nullptr, nullptr, HID, 5, 19, 0, stream);
    k_relu<<<(unsigned)ew_blocks(MH), 256, 0, stream>>>(inp, msg, MH);

    // -------- message passing iterations (DEPTH-1 = 2) --------
    for (int it = 0; it < 2; ++it) {
        k_cvt_pad<<<(unsigned)ew_blocks((size_t)MATOMS * 320), 256, 0, stream>>>(msg, msgbf, MATOMS, HID, 320);
        gemm_launch(msgbf, WqkP, qk, nullptr, nullptr, HID, 10, 19, 0, stream);
        gemm_launch(msgbf, WqbP, qb, nullptr, nullptr, BFD, 10, 1,  0, stream);
        gemm_launch(msgbf, WvP,  vv, nullptr, nullptr, HID, 10, 19, 0, stream);
        k_attn<<<(MATOMS + 7) / 8, 256, 0, stream>>>(msg, qk, qb, vv, inp, f_bonds,
                                                     a2a, a2b, ln_g, ln_b, mwbuf);
        k_cvt_pad<<<(unsigned)ew_blocks((size_t)MATOMS * 320), 256, 0, stream>>>(mwbuf, mwbf, MATOMS, HID, 320);
        gemm_launch(mwbf,  WihP, gi, gru_bih, nullptr, 3 * HID, 10, 57, 0, stream);
        gemm_launch(msgbf, WhhP, gh, gru_bhh, nullptr, 3 * HID, 10, 57, 0, stream);
        k_gru<<<(unsigned)ew_blocks(MH), 256, 0, stream>>>(gi, gh, msg);
    }

    // -------- readout --------
    k_gather_sum<<<(unsigned)ew_blocks(MH), 256, 0, stream>>>(msg, a2a, amsg);
    k_cvt_pad<<<(unsigned)ew_blocks((size_t)MATOMS * 320), 256, 0, stream>>>(amsg, amsgbf, MATOMS, HID, 320);
    gemm_launch(fabf,   WoTP, t_buf, nullptr, nullptr, HID, 5,  19, 0, stream);
    gemm_launch(amsgbf, WoBP, ah,    b_o,     t_buf,   HID, 10, 19, 1, stream);
    k_segmean<<<(unsigned)ew_blocks((size_t)NMOLS * HID), 256, 0, stream>>>(ah, out);
}
